// QuantizedCodebook_7052336300191
// MI455X (gfx1250) — compile-verified, matrix operand order checked by reading, not run on hardware
//
#include <hip/hip_runtime.h>

typedef __attribute__((ext_vector_type(16))) __bf16 v16bf;
typedef __attribute__((ext_vector_type(8)))  float  v8f;

#define N_ROWS   131072      // 128*32*32
#define DIM      64
#define K_CODES  512
#define KT_TILES 32          // 512 / 16
#define FRAG_US  16          // bf16 values per lane per fragment
#define B_FRAG_TOTAL (KT_TILES * 2 * 32 * FRAG_US)   // 32768 ushorts = 64 KB
#define LDS_UINT4 (B_FRAG_TOTAL / 8 + K_CODES / 4)   // 4096 + 128 = 4224

__device__ __forceinline__ unsigned short f2bf_rne(float f) {
  unsigned int u = __float_as_uint(f);
  u += 0x7FFFu + ((u >> 16) & 1u);
  return (unsigned short)(u >> 16);
}

union V16BF {
  v16bf v;
  unsigned short u[16];
  uint4 q[2];
};

// ---------------------------------------------------------------------------
// Prep: pack codebook (512x64 fp32) into bf16 WMMA B-fragments + code_sqr.
// B fragment layout mirrors the ISA 16-bit operand layout:
//   lanes 0-15  : column n = lane,    K = {kb+0..7,  kb+16..23}
//   lanes 16-31 : column n = lane-16, K = {kb+8..15, kb+24..31}
// Workspace layout (contiguous, copied to LDS as one block):
//   [0, 64KB)      bf16 B fragments
//   [64KB, 66KB)   code_sqr (fp32, 512 entries)
// ---------------------------------------------------------------------------
__global__ void vq_prep(const float* __restrict__ cb,
                        unsigned short* __restrict__ wsbf,
                        float* __restrict__ csq) {
  int t = blockIdx.x * blockDim.x + threadIdx.x;   // 0..2047
  int kt   = t >> 6;
  int s    = (t >> 5) & 1;
  int lane = t & 31;
  int col  = lane & 15;
  int half = lane >> 4;
  int code = kt * 16 + col;
  int kb   = s * 32;
  const float* row = cb + code * DIM;
  unsigned short* dst = wsbf + ((kt * 2 + s) * 32 + lane) * FRAG_US;
#pragma unroll
  for (int j = 0; j < 16; ++j) {
    int k = kb + (half ? ((j < 8) ? (j + 8) : (j + 16))
                       : ((j < 8) ?  j      : (j + 8)));
    dst[j] = f2bf_rne(row[k]);
  }
  if (t < K_CODES) {
    const float* r2 = cb + t * DIM;
    float s2 = 0.f;
#pragma unroll
    for (int k = 0; k < DIM; ++k) { float v = r2[k]; s2 += v * v; }
    csq[t] = s2;
  }
}

// ---------------------------------------------------------------------------
// Main: 8 waves/block, each wave owns a 16-row tile of x.
// score_k = ||c_k||^2 - 2 x.c_k computed ENTIRELY inside the WMMA:
//   A = bf16(-2x), C initialized to ||c_k||^2 (column == lane constant)
//   D = A*B + C  ==  score.  Argmin needs only a compare per row.
// Tile loop is ping-pong double-buffered (kt += 2): no rotation movs and
// two independent WMMA chains per iteration to hide WMMA->VALU hazards.
// ---------------------------------------------------------------------------
__global__ void __launch_bounds__(256) vq_main(
    const float* __restrict__ x, const float* __restrict__ cb,
    const uint4* __restrict__ ws, float* __restrict__ out) {
  __shared__ uint4 sh[LDS_UINT4];                    // 66 KB of 320 KB/WGP

  // Stage B fragments + code_sqr into LDS in one pass.
  for (int i = threadIdx.x; i < LDS_UINT4; i += 256) sh[i] = ws[i];
  __syncthreads();

  const unsigned short* ldsb = (const unsigned short*)sh;
  const float* ldsc = (const float*)(sh + B_FRAG_TOTAL / 8);

  const int wave = threadIdx.x >> 5;
  const int lane = threadIdx.x & 31;
  const int mt   = blockIdx.x * 8 + wave;            // 0..8191 row tiles
  const int R    = mt * 16;
  const int row  = lane & 15;
  const int col  = lane & 15;
  const int half = lane >> 4;

  // Build the two A fragments: bf16(-2 * x), K=[0,32) and [32,64).
  V16BF a0, a1;
  {
    const float* rp = x + (size_t)(R + row) * DIM + half * 8;
#pragma unroll
    for (int s = 0; s < 2; ++s) {
      const float* p = rp + s * 32;
      float4 f0 = ((const float4*)p)[0];
      float4 f1 = ((const float4*)p)[1];
      float4 f2 = ((const float4*)(p + 16))[0];
      float4 f3 = ((const float4*)(p + 16))[1];
      V16BF& A = s ? a1 : a0;
      A.u[0]  = f2bf_rne(-2.f * f0.x); A.u[1]  = f2bf_rne(-2.f * f0.y);
      A.u[2]  = f2bf_rne(-2.f * f0.z); A.u[3]  = f2bf_rne(-2.f * f0.w);
      A.u[4]  = f2bf_rne(-2.f * f1.x); A.u[5]  = f2bf_rne(-2.f * f1.y);
      A.u[6]  = f2bf_rne(-2.f * f1.z); A.u[7]  = f2bf_rne(-2.f * f1.w);
      A.u[8]  = f2bf_rne(-2.f * f2.x); A.u[9]  = f2bf_rne(-2.f * f2.y);
      A.u[10] = f2bf_rne(-2.f * f2.z); A.u[11] = f2bf_rne(-2.f * f2.w);
      A.u[12] = f2bf_rne(-2.f * f3.x); A.u[13] = f2bf_rne(-2.f * f3.y);
      A.u[14] = f2bf_rne(-2.f * f3.z); A.u[15] = f2bf_rne(-2.f * f3.w);
    }
  }

  float best[8];
  int   bidx[8];
#pragma unroll
  for (int r = 0; r < 8; ++r) { best[r] = 3.4e38f; bidx[r] = 0; }

#define LOAD_TILE(kt_, F0, F1)                                                 \
  {                                                                            \
    const uint4* p0_ = (const uint4*)&ldsb[(((kt_)*2 + 0) * 32 + lane) * FRAG_US]; \
    const uint4* p1_ = (const uint4*)&ldsb[(((kt_)*2 + 1) * 32 + lane) * FRAG_US]; \
    F0.q[0] = p0_[0]; F0.q[1] = p0_[1];                                        \
    F1.q[0] = p1_[0]; F1.q[1] = p1_[1];                                        \
  }

  V16BF b0e, b1e, b0o, b1o;                          // even / odd tile buffers
  LOAD_TILE(0, b0e, b1e);
  float csE = ldsc[col];
  float csO;

  for (int kt = 0; kt < KT_TILES; kt += 2) {
    // prefetch odd tile kt+1 while computing even tile kt
    LOAD_TILE(kt + 1, b0o, b1o);
    csO = ldsc[(kt + 1) * 16 + col];

    v8f c = {csE, csE, csE, csE, csE, csE, csE, csE};
    c = __builtin_amdgcn_wmma_f32_16x16x32_bf16(false, a0.v, false, b0e.v,
                                                (short)0, c, false, false);
    c = __builtin_amdgcn_wmma_f32_16x16x32_bf16(false, a1.v, false, b1e.v,
                                                (short)0, c, false, false);
    const int codeE = kt * 16 + col;
#pragma unroll
    for (int r = 0; r < 8; ++r) {
      if (c[r] < best[r]) { best[r] = c[r]; bidx[r] = codeE; }
    }

    // prefetch even tile kt+2 (branchless wrap on the last iteration)
    const int ktn = (kt + 2) & (KT_TILES - 1);
    LOAD_TILE(ktn, b0e, b1e);
    csE = ldsc[ktn * 16 + col];

    v8f d = {csO, csO, csO, csO, csO, csO, csO, csO};
    d = __builtin_amdgcn_wmma_f32_16x16x32_bf16(false, a0.v, false, b0o.v,
                                                (short)0, d, false, false);
    d = __builtin_amdgcn_wmma_f32_16x16x32_bf16(false, a1.v, false, b1o.v,
                                                (short)0, d, false, false);
    const int codeO = (kt + 1) * 16 + col;
#pragma unroll
    for (int r = 0; r < 8; ++r) {
      if (d[r] < best[r]) { best[r] = d[r]; bidx[r] = codeO; }
    }
  }
#undef LOAD_TILE

  // Min-reduce across the 16 lanes of each half (xor < 16 stays in-half).
#pragma unroll
  for (int m = 1; m < 16; m <<= 1) {
#pragma unroll
    for (int r = 0; r < 8; ++r) {
      float ob = __shfl_xor(best[r], m, 32);
      int   oi = __shfl_xor(bidx[r], m, 32);
      if (ob < best[r] || (ob == best[r] && oi < bidx[r])) {
        best[r] = ob; bidx[r] = oi;
      }
    }
  }

  float* qout = out + 1;
  float* iout = out + 1 + (size_t)N_ROWS * DIM;

  // C/D layout: lanes 0-15 hold rows 0-7, lanes 16-31 hold rows 8-15.
  if (lane == 0) {
#pragma unroll
    for (int r = 0; r < 8; ++r) iout[R + r] = (float)bidx[r];
  } else if (lane == 16) {
#pragma unroll
    for (int r = 0; r < 8; ++r) iout[R + 8 + r] = (float)bidx[r];
  }

  // Broadcast winning index per row to the whole wave.
  int rowIdx[16];
#pragma unroll
  for (int r = 0; r < 8; ++r) rowIdx[r]     = __shfl(bidx[r], 0, 32);
#pragma unroll
  for (int r = 0; r < 8; ++r) rowIdx[8 + r] = __shfl(bidx[r], 16, 32);

  // Gather codebook rows (exact fp32), write quantize, fuse loss accumulation.
  float lsum = 0.f;
#pragma unroll
  for (int m = 0; m < 16; ++m) {
    int idx = rowIdx[m];
    const float* cr = cb + (size_t)idx * DIM;
    const float* xr = x + (size_t)(R + m) * DIM;
    float q0 = cr[lane],      q1 = cr[lane + 32];
    float x0 = xr[lane],      x1 = xr[lane + 32];
    qout[(size_t)(R + m) * DIM + lane]      = q0;
    qout[(size_t)(R + m) * DIM + lane + 32] = q1;
    float d0 = q0 - x0, d1 = q1 - x1;
    lsum += d0 * d0 + d1 * d1;
  }

#pragma unroll
  for (int m = 1; m < 32; m <<= 1) lsum += __shfl_xor(lsum, m, 32);
  if (lane == 0) {
    const float scale = 1.25f / (float)((size_t)N_ROWS * DIM);
    atomicAdd(out, lsum * scale);
  }
}

extern "C" void kernel_launch(void* const* d_in, const int* in_sizes, int n_in,
                              void* d_out, int out_size, void* d_ws, size_t ws_size,
                              hipStream_t stream) {
  const float* inputs   = (const float*)d_in[0];   // [131072, 64] fp32
  const float* codebook = (const float*)d_in[1];   // [512, 64] fp32
  float* out = (float*)d_out;                      // loss | quantize | indices

  unsigned short* ws_bf = (unsigned short*)d_ws;                    // 64 KB
  float* ws_csq = (float*)((char*)d_ws + (size_t)B_FRAG_TOTAL * 2); // 2 KB

  hipMemsetAsync(out, 0, sizeof(float), stream);   // zero the loss accumulator
  vq_prep<<<8, 256, 0, stream>>>(codebook, ws_bf, ws_csq);
  vq_main<<<1024, 256, 0, stream>>>(inputs, codebook, (const uint4*)d_ws, out);
}
